// CViT_43164421324982
// MI455X (gfx1250) — matrix operand order repaired and checked by reading, not compiled, not run on hardware
//
#include <hip/hip_runtime.h>

typedef _Float16 v16h __attribute__((ext_vector_type(16)));
typedef _Float16 h8   __attribute__((ext_vector_type(8)));
typedef _Float16 h2   __attribute__((ext_vector_type(2)));
typedef float    v8f  __attribute__((ext_vector_type(8)));
typedef unsigned int u32x4 __attribute__((ext_vector_type(4)));
typedef int          i32x4 __attribute__((ext_vector_type(4)));
typedef int          i32x8 __attribute__((ext_vector_type(8)));

#define CVIT_EPS   100000.0f
#define CVIT_G     65536      // 64*64*16 grid points (K dimension)
#define CVIT_C     128
#define CVIT_B     2
#define CVIT_N     4096
#define CVIT_KSTEP 64                      // staged K per step (2 WMMA chunks)
#define CVIT_NSTEP (CVIT_G / CVIT_KSTEP)   // 1024
#define CVIT_XPITCH 72                     // halves per LDS tile row (144 B)

// dynamic LDS layout (bytes):
//   EHs 16896 | EWs 16896 | EZs 6144 | XT0 256*72*2=36864 | XT1 36864
#define CVIT_SMEM  113664

#if defined(__has_builtin)
#if __has_builtin(__builtin_amdgcn_tensor_load_to_lds)
#define CVIT_HAVE_TDM 1
#endif
#endif
#ifndef CVIT_HAVE_TDM
#define CVIT_HAVE_TDM 0
#endif

// ---------------------------------------------------------------------------
// Pass 1: x (B, G, C) f32  ->  xT ((B*C), G) f16   (transpose + convert)
// ---------------------------------------------------------------------------
__global__ __launch_bounds__(256) void cvit_transpose_f16(
    const float* __restrict__ x, _Float16* __restrict__ xT) {
  __shared__ _Float16 tile[64 * 72];
  const int g0  = blockIdx.x * 64;
  const int c0  = blockIdx.y * 64;
  const int bb  = blockIdx.z;
  const int tid = threadIdx.x;
#pragma unroll
  for (int i = 0; i < 16; ++i) {
    int idx = tid + i * 256;
    int cc = idx & 63, gg = idx >> 6;            // coalesced over c
    float v = x[((size_t)bb * CVIT_G + (size_t)(g0 + gg)) * CVIT_C + (c0 + cc)];
    tile[cc * 72 + gg] = (_Float16)v;
  }
  __syncthreads();
#pragma unroll
  for (int i = 0; i < 16; ++i) {
    int idx = tid + i * 256;
    int gg = idx & 63, cc = idx >> 6;            // coalesced over g
    xT[((size_t)bb * CVIT_C + (size_t)(c0 + cc)) * CVIT_G + (g0 + gg)] =
        tile[cc * 72 + gg];
  }
}

// ---------------------------------------------------------------------------
// TDM: one descriptor moves the whole 256x64-element K-tile global->LDS,
// with hardware LDS padding (32 dwords data + 4 dwords pad = 144 B row pitch).
// ---------------------------------------------------------------------------
__device__ __forceinline__ void cvit_tdm_load(unsigned lds_base,
                                              unsigned long long gaddr) {
#if CVIT_HAVE_TDM
  u32x4 g0;
  g0.x = 1u;                                        // count=1 (valid user D#)
  g0.y = lds_base;                                  // lds_addr (bytes)
  g0.z = (unsigned)gaddr;                           // global_addr[31:0]
  g0.w = (unsigned)((gaddr >> 32) & 0x01FFFFFFu)    // global_addr[56:32]
         | 0x80000000u;                             // type=2 ("image")
  i32x8 g1;
  g1[0] = (int)((1u << 16)      // data_size = 2 B
                | (1u << 20)    // pad_enable
                | (4u << 22)    // pad_interval: 32 dwords
                | (3u << 25));  // pad_amount:   4 dwords (16 B)
  g1[1] = 0;                    // atomic_barrier off; tensor_dim0 low16 = 0
  g1[2] = (int)((256u << 16) | 1u);   // tensor_dim1 lo16=256 | dim0 hi16=1
  g1[3] = (int)(64u << 16);           // tile_dim0=64 | tensor_dim1 hi16=0
  g1[4] = 256;                        // tile_dim1=256, tile_dim2=0
  g1[5] = 0x10000;                    // tensor_dim0_stride = 65536 (lo32)
  g1[6] = 0;
  g1[7] = 0;
  i32x4 z4 = {0, 0, 0, 0};
#if __clang_major__ >= 23
  i32x8 z8 = {0, 0, 0, 0, 0, 0, 0, 0};
  __builtin_amdgcn_tensor_load_to_lds(g0, g1, z4, z4, z8, 0);
#else
  __builtin_amdgcn_tensor_load_to_lds(g0, g1, z4, z4, 0);
#endif
#endif
}

// ---------------------------------------------------------------------------
// Fallback CDNA5 async copy: global -> LDS, 128 B per lane (8 x B128)
// ---------------------------------------------------------------------------
__device__ __forceinline__ void cvit_async_row(unsigned lds_row, unsigned gofs,
                                               unsigned long long gbase) {
  asm volatile(
      "global_load_async_to_lds_b128 %0, %1, %2 offset:0\n\t"
      "global_load_async_to_lds_b128 %0, %1, %2 offset:16\n\t"
      "global_load_async_to_lds_b128 %0, %1, %2 offset:32\n\t"
      "global_load_async_to_lds_b128 %0, %1, %2 offset:48\n\t"
      "global_load_async_to_lds_b128 %0, %1, %2 offset:64\n\t"
      "global_load_async_to_lds_b128 %0, %1, %2 offset:80\n\t"
      "global_load_async_to_lds_b128 %0, %1, %2 offset:96\n\t"
      "global_load_async_to_lds_b128 %0, %1, %2 offset:112"
      :: "v"(lds_row), "v"(gofs), "s"(gbase) : "memory");
}

__device__ __forceinline__ void cvit_wait_stage() {
#if CVIT_HAVE_TDM
  __builtin_amdgcn_s_wait_tensorcnt(0);
#else
  asm volatile("s_wait_asynccnt 0x0" ::: "memory");
#endif
}

__device__ __forceinline__ float cvit_near2(float c, float scale) {
  float g = floorf(c * scale + 0.5f);
  g = fminf(fmaxf(g, 0.0f), scale);
  float t = c - g * (1.0f / scale);
  return t * t;    // squared distance to nearest grid plane (softmax shift)
}

// ---------------------------------------------------------------------------
// Pass 2: fused separable-RBF weight generation + WMMA GEMM
//   A (16x32 f16) = weights generated in registers from LDS tables
//   B (32x16 f16) = xT K-tile, double-buffered in LDS via TDM (or async)
// Block: 128 queries x 256 (b,c) columns; K streamed 64 wide (2 chunks/step).
// ---------------------------------------------------------------------------
__global__ __launch_bounds__(256) void cvit_main(
    const float* __restrict__ coords, const _Float16* __restrict__ xT,
    float* __restrict__ out) {
  extern __shared__ __align__(16) char smem[];
  _Float16* EHs = (_Float16*)smem;            // [128][66]
  _Float16* EWs = EHs + 128 * 66;             // [128][66]
  _Float16* EZs = EWs + 128 * 66;             // [128][24] (invS folded in)
  _Float16* XT0 = EZs + 128 * 24;             // [256][72] K-tile buf 0
  _Float16* XT1 = XT0 + 256 * CVIT_XPITCH;    // [256][72] K-tile buf 1

  const int tid = threadIdx.x;
  const int q0  = blockIdx.x * 128;

  // ---- per-query stability-shifted 1-D RBF tables (every entry <= 1) ----
  if (tid < 128) {
    const int q = q0 + tid;
    const float cx = coords[3 * q + 0];
    const float cy = coords[3 * q + 1];
    const float cz = coords[3 * q + 2];
    const float mx = cvit_near2(cx, 63.0f);
    const float my = cvit_near2(cy, 63.0f);
    const float mz = cvit_near2(cz, 15.0f);
    float sx = 0.0f, sy = 0.0f, sz = 0.0f;
    for (int i = 0; i < 64; ++i) {
      float t = cx - (float)i * (1.0f / 63.0f);
      float e = __expf(-CVIT_EPS * (t * t - mx));
      sx += e;
      EHs[tid * 66 + i] = (_Float16)e;
    }
    for (int i = 0; i < 64; ++i) {
      float t = cy - (float)i * (1.0f / 63.0f);
      float e = __expf(-CVIT_EPS * (t * t - my));
      sy += e;
      EWs[tid * 66 + i] = (_Float16)e;
    }
    float ezv[16];
    for (int i = 0; i < 16; ++i) {
      float t = cz - (float)i * (1.0f / 15.0f);
      float e = __expf(-CVIT_EPS * (t * t - mz));
      sz += e;
      ezv[i] = e;
    }
    const float inv = 1.0f / (sx * sy * sz);   // <= 1, fold into Ez
    for (int i = 0; i < 16; ++i)
      EZs[tid * 24 + i] = (_Float16)(ezv[i] * inv);
  }

  const int lane = tid & 31;
  const int wave = tid >> 5;
  const int mg   = wave & 1;        // query half: rows mg*64 .. mg*64+63
  const int cgrp = wave >> 1;       // column group: cols cgrp*64 .. +63
  const int lr   = lane & 15;
  const int koff = (lane & 16) ? 8 : 0;    // A-layout: ez sub-row select
  const int kbo  = (lane & 16) ? 8 : 0;    // B-layout: k octet select

  v8f acc[4][4] = {};   // 4 m-subtiles x 4 n-subtiles of 16x16 f32

  const unsigned long long gb = (unsigned long long)(uintptr_t)xT;
  unsigned lb_cur  = (unsigned)(uintptr_t)XT0;   // TDM LDS dest (buffer base)
  unsigned lb_next = (unsigned)(uintptr_t)XT1;
#if !CVIT_HAVE_TDM
  const unsigned gofs0 = (unsigned)tid * (unsigned)(CVIT_G * 2);
  unsigned lds_cur  = (unsigned)(uintptr_t)(XT0 + tid * CVIT_XPITCH);
  unsigned lds_next = (unsigned)(uintptr_t)(XT1 + tid * CVIT_XPITCH);
#endif

  // stage tile 0 -> buf 0
#if CVIT_HAVE_TDM
  if (wave == 0) cvit_tdm_load(lb_cur, gb);
#else
  cvit_async_row(lds_cur, gofs0, gb);
#endif
  cvit_wait_stage();
  __syncthreads();                           // tables + tile 0 ready

  const _Float16* cbuf = XT0;
  const _Float16* obuf = XT1;

#pragma unroll 1
  for (int s = 0; s < CVIT_NSTEP; ++s) {
    if (s + 1 < CVIT_NSTEP) {                // stage next K-tile
#if CVIT_HAVE_TDM
      if (wave == 0)
        cvit_tdm_load(lb_next, gb + (unsigned long long)(s + 1) * 128ull);
#else
      cvit_async_row(lds_next, gofs0 + (unsigned)(s + 1) * 128u, gb);
#endif
    }

    const int gbase0 = s * CVIT_KSTEP;
    const int hh = gbase0 >> 10;             // constant over 64-wide K tile
    const int w0 = (gbase0 >> 4) & 63;       // spans w0 .. w0+3

    // step-invariant table values (reused by both k-chunks)
    _Float16 ehv[4];
    h8 ezvv[4];
#pragma unroll
    for (int mi = 0; mi < 4; ++mi) {
      const int rr = (mg * 4 + mi) * 16 + lr;
      ehv[mi]  = EHs[rr * 66 + hh];
      ezvv[mi] = *(const h8*)&EZs[rr * 24 + koff];   // one 16B LDS load
    }

#pragma unroll
    for (int kc = 0; kc < 2; ++kc) {
      const int w0c = w0 + 2 * kc;

      // ---- A fragments (weights) in registers, packed f16 ----
      v16h Af[4];
#pragma unroll
      for (int mi = 0; mi < 4; ++mi) {
        const int rr = (mg * 4 + mi) * 16 + lr;
        h2 ew = *(const h2*)&EWs[rr * 66 + w0c];
        h8 loh = ezvv[mi] * (_Float16)(ehv[mi] * ew.x);   // v_pk_mul_f16
        h8 hih = ezvv[mi] * (_Float16)(ehv[mi] * ew.y);
        Af[mi] = __builtin_shufflevector(loh, hih, 0, 1, 2, 3, 4, 5, 6, 7, 8,
                                         9, 10, 11, 12, 13, 14, 15);
      }

      // ---- B fragments for this k-chunk ----
      v16h Bf[4];
#pragma unroll
      for (int ni = 0; ni < 4; ++ni) {
        const _Float16* bp =
            cbuf + (cgrp * 64 + ni * 16 + lr) * CVIT_XPITCH + kc * 32 + kbo;
        h8 b0v = *(const h8*)bp;
        h8 b1v = *(const h8*)(bp + 16);
        Bf[ni] = __builtin_shufflevector(b0v, b1v, 0, 1, 2, 3, 4, 5, 6, 7, 8,
                                         9, 10, 11, 12, 13, 14, 15);
      }

      // ---- 16 WMMAs per chunk (32 per barrier) ----
#pragma unroll
      for (int ni = 0; ni < 4; ++ni) {
#pragma unroll
        for (int mi = 0; mi < 4; ++mi) {
          acc[mi][ni] = __builtin_amdgcn_wmma_f32_16x16x32_f16(
              false, Af[mi], false, Bf[ni], (short)0, acc[mi][ni], false,
              false);
        }
      }
    }

    cvit_wait_stage();                       // next tile landed in LDS
    __syncthreads();                         // single barrier per K-step
    const _Float16* tp = cbuf; cbuf = obuf; obuf = tp;
    unsigned tb = lb_cur; lb_cur = lb_next; lb_next = tb;
#if !CVIT_HAVE_TDM
    unsigned tl = lds_cur; lds_cur = lds_next; lds_next = tl;
#endif
  }

  // ---- epilogue: D layout = VGPR r -> (M=r | r+8), lanes -> N column ----
  const int mup = (lane & 16) ? 8 : 0;
#pragma unroll
  for (int mi = 0; mi < 4; ++mi) {
#pragma unroll
    for (int ni = 0; ni < 4; ++ni) {
      v8f a = acc[mi][ni];
      const int q  = q0 + mg * 64 + mi * 16 + mup;
      const int bc = cgrp * 64 + ni * 16 + lr;
      const int ob = bc >> 7;
      const int oc = bc & 127;
      float* op = out + ((size_t)ob * CVIT_N + (size_t)q) * CVIT_C + oc;
#pragma unroll
      for (int r = 0; r < 8; ++r) op[(size_t)r * CVIT_C] = a[r];
    }
  }
}

// ---------------------------------------------------------------------------
extern "C" void kernel_launch(void* const* d_in, const int* in_sizes, int n_in,
                              void* d_out, int out_size, void* d_ws,
                              size_t ws_size, hipStream_t stream) {
  const float* x      = (const float*)d_in[0];   // (B, H, W, D, C) f32
  const float* coords = (const float*)d_in[1];   // (N, 3) f32
  float* out          = (float*)d_out;           // (B, N, C) f32
  _Float16* xT        = (_Float16*)d_ws;         // 32 MB scratch: (B*C, G) f16

  (void)hipFuncSetAttribute((const void*)cvit_main,
                            hipFuncAttributeMaxDynamicSharedMemorySize,
                            CVIT_SMEM);

  cvit_transpose_f16<<<dim3(CVIT_G / 64, CVIT_C / 64, CVIT_B), 256, 0,
                       stream>>>(x, xT);
  cvit_main<<<dim3(CVIT_N / 128), 256, CVIT_SMEM, stream>>>(coords, xT, out);
}
